// CNNModel_79035988181541
// MI455X (gfx1250) — matrix operand order repaired and checked by reading, not compiled, not run on hardware
//
#include <hip/hip_runtime.h>
#include <hip/hip_bf16.h>
#include <math.h>

// ---------------------------------------------------------------------------
// Types for CDNA5 WMMA (wave32, v_wmma_f32_16x16x32_bf16)
// ---------------------------------------------------------------------------
typedef __attribute__((ext_vector_type(16))) __bf16          v16bf;
typedef __attribute__((ext_vector_type(16))) unsigned short  v16u;
typedef __attribute__((ext_vector_type(8)))  unsigned short  v8u;
typedef __attribute__((ext_vector_type(8)))  float           v8f;

__device__ __forceinline__ unsigned short f2bf(float f) {
    unsigned int x = __float_as_uint(f);
    x += 0x7fffu + ((x >> 16) & 1u);       // round-to-nearest-even
    return (unsigned short)(x >> 16);
}
__device__ __forceinline__ float sigmf(float x) { return 1.0f / (1.0f + __expf(-x)); }

// A-fragment (16x32 bf16, one 16x16x32 tile): lane L holds row M=L&15.
// lanes 0-15: elements 0..7 => K=k0..k0+7,   8..15 => K=k0+16..k0+23
// lanes 16-31: same rows, K offset +8 (kb).  p points at (row, k0+kb).
__device__ __forceinline__ v16bf ld_frag_split(const unsigned short* p) {
    v8u a = *(const v8u*)p;
    v8u b = *(const v8u*)(p + 16);
    v16u u = __builtin_shufflevector(a, b, 0,1,2,3,4,5,6,7,8,9,10,11,12,13,14,15);
    union { v16u u; v16bf b; } cv; cv.u = u; return cv.b;
}
// B-fragment (32x16 bf16): lane L holds column N=L&15, elements 0..15 => 16 K values,
// i.e. 16 contiguous bf16 of a n-major weight row.
__device__ __forceinline__ v16bf ld_frag_contig(const unsigned short* p) {
    union { v16u u; v16bf b; } cv; cv.u = *(const v16u*)p; return cv.b;
}

// ---------------------------------------------------------------------------
// Model constants
// ---------------------------------------------------------------------------
#define Bsz 16
#define Tsz 1024
#define Fsz 128
#define HID 256
#define OUTC 61
#define GW  768   // 3*HID

// ---------------------------------------------------------------------------
// Conv front-end: conv1 + BN + ReLU + maxpool2 fused (recompute 4 conv taps
// per pooled output instead of materializing a 65 MB intermediate).
// in: (16,1,1024,128)  w: (8,1,K,K)  out: (16,8,HP,WP)
// ---------------------------------------------------------------------------
__global__ void conv1_bn_relu_pool(const float* __restrict__ in,
                                   const float* __restrict__ w,
                                   const float* __restrict__ bias,
                                   const float* __restrict__ bg, const float* __restrict__ bb,
                                   const float* __restrict__ bm, const float* __restrict__ bv,
                                   float* __restrict__ out,
                                   int K, int S, int HP, int WP)
{
    int idx = blockIdx.x * blockDim.x + threadIdx.x;
    int total = Bsz * 8 * HP * WP;
    if (idx >= total) return;
    int wp = idx % WP; int t = idx / WP;
    int hp = t % HP;   t /= HP;
    int c  = t % 8;    int b = t / 8;

    float scale = bg[c] * rsqrtf(bv[c] + 1e-5f);
    float beta  = bb[c], mean = bm[c], bs = bias[c];
    const float* wr  = w + c * K * K;
    const float* inb = in + (size_t)b * Tsz * Fsz;

    float mx = -3.4e38f;
    for (int dy = 0; dy < 2; ++dy) {
        for (int dx = 0; dx < 2; ++dx) {
            int hi = (2 * hp + dy) * S, wi = (2 * wp + dx) * S;
            float acc = bs;
            for (int i = 0; i < K; ++i)
                for (int j = 0; j < K; ++j)
                    acc += inb[(hi + i) * Fsz + wi + j] * wr[i * K + j];
            float v = fmaxf((acc - mean) * scale + beta, 0.0f);
            mx = fmaxf(mx, v);
        }
    }
    out[idx] = mx;
}

// conv2 + BN + ReLU : in (16,8,Hin,Win) -> out (16,3,H2,W2), weight (3,8,K,K)
__global__ void conv2_bn_relu(const float* __restrict__ in,
                              const float* __restrict__ w,
                              const float* __restrict__ bias,
                              const float* __restrict__ bg, const float* __restrict__ bb,
                              const float* __restrict__ bm, const float* __restrict__ bv,
                              float* __restrict__ out,
                              int Hin, int Win, int K, int S, int H2, int W2)
{
    int idx = blockIdx.x * blockDim.x + threadIdx.x;
    int total = Bsz * 3 * H2 * W2;
    if (idx >= total) return;
    int wo = idx % W2; int t = idx / W2;
    int ho = t % H2;   t /= H2;
    int c  = t % 3;    int b = t / 3;

    float scale = bg[c] * rsqrtf(bv[c] + 1e-5f);
    float acc = bias[c];
    for (int ci = 0; ci < 8; ++ci) {
        const float* ib = in + ((size_t)(b * 8 + ci) * Hin + ho * S) * Win + wo * S;
        const float* wr = w + ((size_t)c * 8 + ci) * K * K;
        for (int i = 0; i < K; ++i)
            for (int j = 0; j < K; ++j)
                acc += ib[i * Win + j] * wr[i * K + j];
    }
    out[idx] = fmaxf((acc - bm[c]) * scale + bb[c], 0.0f);
}

// bilinear resize (reference semantics) writing into the concatenated feats.
__global__ void resize_to_feats(const float* __restrict__ r2,
                                const int* __restrict__ lengths,
                                float* __restrict__ feats,
                                int H, int W, int Ho, int Wo, int segOff)
{
    int idx = blockIdx.x * blockDim.x + threadIdx.x;
    int total = Bsz * 3 * Ho * Wo;
    if (idx >= total) return;
    int wo = idx % Wo; int t = idx / Wo;
    int ho = t % Ho;   t /= Ho;
    int c  = t % 3;    int b = t / 3;

    int ml = 1;
    for (int i = 0; i < Bsz; ++i) ml = max(ml, lengths[i]);
    int hs = (int)((float)H * ((float)lengths[b] / (float)ml));

    float cc = ((float)ho + 0.5f) * ((float)hs / (float)Ho) - 0.5f;
    cc = fmaxf(cc, 0.0f);
    int himax = hs - 1;
    int i0 = min((int)floorf(cc), himax); if (i0 < 0) i0 = 0;
    int i1 = min(i0 + 1, himax);          if (i1 < 0) i1 = 0;
    float w1 = fminf(fmaxf(cc - (float)i0, 0.0f), 1.0f);

    float cw = ((float)wo + 0.5f) * ((float)W / (float)Wo) - 0.5f;
    cw = fminf(fmaxf(cw, 0.0f), (float)(W - 1));
    int j0 = (int)floorf(cw);
    int j1 = min(j0 + 1, W - 1);
    float ww = cw - (float)j0;

    const float* base = r2 + ((size_t)(b * 3 + c) * H) * W;
    float v0 = base[i0 * W + j0] * (1.f - w1) + base[i1 * W + j0] * w1;
    float v1 = base[i0 * W + j1] * (1.f - w1) + base[i1 * W + j1] * w1;
    feats[(size_t)b * 5040 + segOff + (c * Ho + ho) * Wo + wo] = v0 * (1.f - ww) + v1 * ww;
}

// fc1: rfeat = feats(16x5040) @ fc1_w(512,5040)^T + b
__global__ void fc1_kernel(const float* __restrict__ feats,
                           const float* __restrict__ w,
                           const float* __restrict__ bias,
                           float* __restrict__ rfeat)
{
    int idx = blockIdx.x * blockDim.x + threadIdx.x;
    if (idx >= Bsz * 512) return;
    int o = idx & 511, b = idx >> 9;
    const float* f  = feats + (size_t)b * 5040;
    const float* wr = w + (size_t)o * 5040;
    float acc = bias[o];
    for (int k = 0; k < 5040; ++k) acc += f[k] * wr[k];
    rfeat[idx] = acc;
}

// ---------------------------------------------------------------------------
// bf16 packs
// ---------------------------------------------------------------------------
__global__ void pack_bf16(const float* __restrict__ src, unsigned short* __restrict__ dst, int n)
{
    int i = blockIdx.x * blockDim.x + threadIdx.x;
    if (i < n) dst[i] = f2bf(src[i]);
}
// whh (768,256) -> bf16 (768,256) n-major: B-fragment K runs become contiguous
__global__ void pack_whh_T(const float* __restrict__ whh, unsigned short* __restrict__ dst)
{
    int i = blockIdx.x * blockDim.x + threadIdx.x;
    if (i >= GW * HID) return;
    int k = i & (HID - 1), n = i >> 8;
    dst[(size_t)n * HID + k] = f2bf(whh[(size_t)n * HID + k]);
}
// seq = concat(batch, rfeat broadcast) -> bf16 rows of (b*T+t, 640)
__global__ void seq_build(const float* __restrict__ batch,
                          const float* __restrict__ rfeat,
                          unsigned short* __restrict__ seq)
{
    int i = blockIdx.x * blockDim.x + threadIdx.x;
    if (i >= Bsz * Tsz * 640) return;
    int j = i % 640; int row = i / 640;
    int t = row & (Tsz - 1), b = row >> 10;
    float v = (j < Fsz) ? batch[((size_t)b * Tsz + t) * Fsz + j]
                        : rfeat[(size_t)b * 512 + (j - Fsz)];
    seq[i] = f2bf(v);
}

// ---------------------------------------------------------------------------
// WMMA GEMM: C(MxN,f32) = A(MxK,bf16) @ W(NxK,bf16)^T + bias
// 64x64 tile / 128 threads (4 waves); wave w owns rows 16w..16w+15 x 64 cols.
// M%64==0, N%64==0, K%32==0.
// ---------------------------------------------------------------------------
__global__ void __launch_bounds__(128)
gemm_bf16(const unsigned short* __restrict__ A,
          const unsigned short* __restrict__ W,
          const float* __restrict__ bias,
          float* __restrict__ C, int M, int N, int K)
{
    __shared__ __align__(32) unsigned short As[64 * 32];   // A tile, row-major
    __shared__ __align__(32) unsigned short Wt[32 * 64];   // W tile row-major: [n][k]

    int m0 = blockIdx.x * 64, n0 = blockIdx.y * 64;
    int tid = threadIdx.x, wave = tid >> 5, lane = tid & 31;
    int row = lane & 15;
    int kb  = (lane >> 4) << 3;            // 0 or 8 (A layout K offset per half-wave)

    v8f zero = {0.f,0.f,0.f,0.f,0.f,0.f,0.f,0.f};
    v8f acc[4] = {zero, zero, zero, zero};

    int r = tid >> 1, hh = tid & 1;        // r: 0..63, hh: K-half
    for (int k0 = 0; k0 < K; k0 += 32) {
        __syncthreads();
        // stage A (64x32): one 32B vector load per thread
        *(v16u*)(As + r * 32 + hh * 16) =
            *(const v16u*)(A + (size_t)(m0 + r) * K + k0 + hh * 16);
        // stage W row-major (64x32): vector load, scalar LDS stores
        v16u wv = *(const v16u*)(W + (size_t)(n0 + r) * K + k0 + hh * 16);
        #pragma unroll
        for (int e = 0; e < 16; ++e) Wt[r * 32 + hh * 16 + e] = wv[e];
        __syncthreads();

        v16bf a = ld_frag_split(As + (wave * 16 + row) * 32 + kb);
        #pragma unroll
        for (int nt = 0; nt < 4; ++nt) {
            v16bf bfr = ld_frag_split(Wt + (nt * 16 + row) * 32 + kb);
            acc[nt] = __builtin_amdgcn_wmma_f32_16x16x32_bf16(
                          false, a, false, bfr, (short)0, acc[nt], false, false);
        }
    }

    int mrow = m0 + wave * 16 + ((lane >> 4) << 3);
    #pragma unroll
    for (int nt = 0; nt < 4; ++nt) {
        int n = n0 + nt * 16 + (lane & 15);
        float bv = bias[n];
        #pragma unroll
        for (int i = 0; i < 8; ++i)
            C[(size_t)(mrow + i) * N + n] = acc[nt][i] + bv;
    }
}

// ---------------------------------------------------------------------------
// GRU scan. grid.x = 2 (0: forward, 1: backward). 384 threads = 12 waves.
// Per step: hW = h(16x256) @ whh^T => 48 WMMA N-tiles (4 per wave) x 8 K-steps,
// gate buffer in LDS, then fused gate update. Backward direction indexes xW
// and the output at rev(b,t) directly (rev is an involution on [0,L)).
// ---------------------------------------------------------------------------
__global__ void __launch_bounds__(384)
gru_scan(const float* __restrict__ xWf, const float* __restrict__ xWb,
         const unsigned short* __restrict__ whhTf, const unsigned short* __restrict__ whhTb,
         const float* __restrict__ bhhf, const float* __restrict__ bhhb,
         const int* __restrict__ lengths, float* __restrict__ out)
{
    const int dir = blockIdx.x;
    const float*          xW   = dir ? xWb   : xWf;
    const unsigned short* whhT = dir ? whhTb : whhTf;
    const float*          bhh  = dir ? bhhb  : bhhf;

    __shared__ __align__(32) float          hF[Bsz * HID];   // fp32 state
    __shared__ __align__(32) unsigned short hB[Bsz * HID];   // bf16 mirror (A operand)
    __shared__ __align__(32) float          G[Bsz * GW];     // hW + bhh gate buffer
    __shared__ int lenS[Bsz];

    int tid = threadIdx.x;
    for (int i = tid; i < Bsz * HID; i += 384) { hF[i] = 0.f; hB[i] = 0; }
    if (tid < Bsz) lenS[tid] = lengths[tid];
    __syncthreads();

    int wave = tid >> 5, lane = tid & 31;
    int row  = lane & 15;
    int kb   = (lane >> 4) << 3;
    int mrow = (lane >> 4) << 3;

    for (int t = 0; t < Tsz; ++t) {
        // ---- WMMA phase: G = h @ whh^T + bhh --------------------------------
        for (int nt = 0; nt < 4; ++nt) {
            int nbase = (wave * 4 + nt) * 16;
            v8f acc = {0.f,0.f,0.f,0.f,0.f,0.f,0.f,0.f};
            #pragma unroll
            for (int ks = 0; ks < 8; ++ks) {
                int k0 = ks * 32;
                if (ks < 7)
                    __builtin_prefetch(whhT + (size_t)(k0 + 32 + lane) * GW + nbase, 0, 3);
                v16bf a   = ld_frag_split(hB + row * HID + k0 + kb);
                v16bf bfr = ld_frag_split(whhT + (size_t)(nbase + row) * HID + k0 + kb);
                acc = __builtin_amdgcn_wmma_f32_16x16x32_bf16(
                          false, a, false, bfr, (short)0, acc, false, false);
            }
            int n = nbase + (lane & 15);
            float bb = bhh[n];
            #pragma unroll
            for (int i = 0; i < 8; ++i) G[(mrow + i) * GW + n] = acc[i] + bb;
        }
        __syncthreads();
        // ---- gate update phase ---------------------------------------------
        for (int idx = tid; idx < Bsz * HID; idx += 384) {
            int b = idx >> 8, j = idx & (HID - 1);
            int len  = lenS[b];
            int tpos = (dir == 0) ? t : ((t < len) ? (len - 1 - t) : t);
            const float* xr = xW + ((size_t)b * Tsz + tpos) * GW;
            const float* gb = G + b * GW;
            float rr = sigmf(xr[j]           + gb[j]);
            float zz = sigmf(xr[HID + j]     + gb[HID + j]);
            float nn = tanhf(xr[2 * HID + j] + rr * gb[2 * HID + j]);
            float hn = (1.f - zz) * nn + zz * hF[idx];
            hF[idx] = hn;
            hB[idx] = f2bf(hn);
            out[((size_t)b * Tsz + tpos) * 512 + dir * HID + j] = hn;
        }
        __syncthreads();
    }
}

// ---------------------------------------------------------------------------
// Final FC + length-gather: out[prefix(b)+t] = h1[b,t] @ fc_w^T + fc_b
// ---------------------------------------------------------------------------
__global__ void __launch_bounds__(64)
final_fc_gather(const float* __restrict__ h1, const float* __restrict__ w,
                const float* __restrict__ bias, const int* __restrict__ lengths,
                float* __restrict__ out)
{
    int bt = blockIdx.x;
    int b = bt >> 10, t = bt & (Tsz - 1);
    int len = lengths[b];
    if (t >= len) return;                       // uniform per block
    int off = 0;
    for (int i = 0; i < b; ++i) off += lengths[i];

    __shared__ float hrow[512];
    const float* src = h1 + ((size_t)b * Tsz + t) * 512;
    for (int i = threadIdx.x; i < 512; i += 64) hrow[i] = src[i];
    __syncthreads();

    for (int o = threadIdx.x; o < OUTC; o += 64) {
        const float* wr = w + (size_t)o * 512;
        float acc = bias[o];
        for (int k = 0; k < 512; ++k) acc += hrow[k] * wr[k];
        out[((size_t)off + t) * OUTC + o] = acc;
    }
}

// ---------------------------------------------------------------------------
// Workspace layout (bytes). Conv scratch overlays the xW region: it is fully
// consumed (feats -> fc1 -> rfeat -> seq) before the first GEMM writes xW.
// Layer-1 output reuses the layer-0 output buffer (dead after bf16 pack).
// Peak usage ~216 MB.
// ---------------------------------------------------------------------------
static const size_t WS_SEQ   = 0;                // 16384*640*2  = 20.97 MB
static const size_t WS_XWF   = 24ull  << 20;     // 16384*768*4  = 50.33 MB
static const size_t WS_XWB   = 80ull  << 20;     // 50.33 MB
static const size_t WS_H0    = 136ull << 20;     // 16384*512*4 = 33.55 MB (also h1)
static const size_t WS_H0BF  = 180ull << 20;     // 16384*512*2 = 16.78 MB
static const size_t WS_WIH0F = 208ull << 20;     // 768*640*2
static const size_t WS_WIH0B = 209ull << 20;
static const size_t WS_WIH1F = 210ull << 20;     // 768*512*2
static const size_t WS_WIH1B = 211ull << 20;
static const size_t WS_WHT0F = 212ull << 20;     // 256*768*2
static const size_t WS_WHT0B = 213ull << 20;
static const size_t WS_WHT1F = 214ull << 20;
static const size_t WS_WHT1B = 215ull << 20;
// conv scratch (overlays XWF window, dead before GEMM)
static const size_t WS_R1_4  = 24ull << 20;      // 16,8,510,62
static const size_t WS_R1_8  = 41ull << 20;      // 16,8,254,30
static const size_t WS_R1_16 = 45ull << 20;      // 16,8,126,14
static const size_t WS_R1_32 = 46ull << 20;      // 16,8,62,6
static const size_t WS_R2_4  = 47ull << 20;      // 16,3,169,20
static const size_t WS_R2_8  = 48ull << 20;      // 16,3,126,14
static const size_t WS_R2_16 = 49ull << 20;      // 16,3,126,14
static const size_t WS_R2_32 = 50ull << 20;      // 16,3,62,6
static const size_t WS_FEATS = 51ull << 20;      // 16,5040
static const size_t WS_RFEAT = 52ull << 20;      // 16,512

extern "C" void kernel_launch(void* const* d_in, const int* in_sizes, int n_in,
                              void* d_out, int out_size, void* d_ws, size_t ws_size,
                              hipStream_t stream)
{
    (void)in_sizes; (void)n_in; (void)out_size; (void)ws_size;

    const float* batch = (const float*)d_in[0];
    const float* c1w[4] = {(const float*)d_in[1], (const float*)d_in[3],
                           (const float*)d_in[5], (const float*)d_in[7]};
    const float* c1b[4] = {(const float*)d_in[2], (const float*)d_in[4],
                           (const float*)d_in[6], (const float*)d_in[8]};
    const float* c2w[4] = {(const float*)d_in[9],  (const float*)d_in[11],
                           (const float*)d_in[13], (const float*)d_in[15]};
    const float* c2b[4] = {(const float*)d_in[10], (const float*)d_in[12],
                           (const float*)d_in[14], (const float*)d_in[16]};
    const float *bn8g = (const float*)d_in[17], *bn8b = (const float*)d_in[18];
    const float *bn8m = (const float*)d_in[19], *bn8v = (const float*)d_in[20];
    const float *bn3g = (const float*)d_in[21], *bn3b = (const float*)d_in[22];
    const float *bn3m = (const float*)d_in[23], *bn3v = (const float*)d_in[24];
    const float *fc1w = (const float*)d_in[25], *fc1b = (const float*)d_in[26];
    const float *g0f_wih = (const float*)d_in[27], *g0f_whh = (const float*)d_in[28];
    const float *g0f_bih = (const float*)d_in[29], *g0f_bhh = (const float*)d_in[30];
    const float *g0b_wih = (const float*)d_in[31], *g0b_whh = (const float*)d_in[32];
    const float *g0b_bih = (const float*)d_in[33], *g0b_bhh = (const float*)d_in[34];
    const float *g1f_wih = (const float*)d_in[35], *g1f_whh = (const float*)d_in[36];
    const float *g1f_bih = (const float*)d_in[37], *g1f_bhh = (const float*)d_in[38];
    const float *g1b_wih = (const float*)d_in[39], *g1b_whh = (const float*)d_in[40];
    const float *g1b_bih = (const float*)d_in[41], *g1b_bhh = (const float*)d_in[42];
    const float *fcw = (const float*)d_in[43], *fcb = (const float*)d_in[44];
    const int* lengths = (const int*)d_in[45];

    char* ws = (char*)d_ws;
    unsigned short* SEQ   = (unsigned short*)(ws + WS_SEQ);
    float*          XWF   = (float*)(ws + WS_XWF);
    float*          XWB   = (float*)(ws + WS_XWB);
    float*          H0    = (float*)(ws + WS_H0);       // layer0 out, reused for layer1 out
    unsigned short* H0BF  = (unsigned short*)(ws + WS_H0BF);
    unsigned short* WIH0F = (unsigned short*)(ws + WS_WIH0F);
    unsigned short* WIH0B = (unsigned short*)(ws + WS_WIH0B);
    unsigned short* WIH1F = (unsigned short*)(ws + WS_WIH1F);
    unsigned short* WIH1B = (unsigned short*)(ws + WS_WIH1B);
    unsigned short* WHT0F = (unsigned short*)(ws + WS_WHT0F);
    unsigned short* WHT0B = (unsigned short*)(ws + WS_WHT0B);
    unsigned short* WHT1F = (unsigned short*)(ws + WS_WHT1F);
    unsigned short* WHT1B = (unsigned short*)(ws + WS_WHT1B);

    // ---- 1) weight packs (bf16, layouts unchanged) --------------------------
    pack_bf16<<<(GW*640 + 255)/256, 256, 0, stream>>>(g0f_wih, WIH0F, GW*640);
    pack_bf16<<<(GW*640 + 255)/256, 256, 0, stream>>>(g0b_wih, WIH0B, GW*640);
    pack_bf16<<<(GW*512 + 255)/256, 256, 0, stream>>>(g1f_wih, WIH1F, GW*512);
    pack_bf16<<<(GW*512 + 255)/256, 256, 0, stream>>>(g1b_wih, WIH1B, GW*512);
    pack_whh_T<<<(GW*HID + 255)/256, 256, 0, stream>>>(g0f_whh, WHT0F);
    pack_whh_T<<<(GW*HID + 255)/256, 256, 0, stream>>>(g0b_whh, WHT0B);
    pack_whh_T<<<(GW*HID + 255)/256, 256, 0, stream>>>(g1f_whh, WHT1F);
    pack_whh_T<<<(GW*HID + 255)/256, 256, 0, stream>>>(g1b_whh, WHT1B);

    // ---- 2) conv front-end -------------------------------------------------
    struct C1 { int K, S, HP, WP; size_t off; };
    const C1 b1[4] = {{4,1,510,62,WS_R1_4}, {8,2,254,30,WS_R1_8},
                      {16,4,126,14,WS_R1_16}, {32,8,62,6,WS_R1_32}};
    for (int i = 0; i < 4; ++i) {
        int total = Bsz * 8 * b1[i].HP * b1[i].WP;
        conv1_bn_relu_pool<<<(total + 255)/256, 256, 0, stream>>>(
            batch, c1w[i], c1b[i], bn8g, bn8b, bn8m, bn8v,
            (float*)(ws + b1[i].off), b1[i].K, b1[i].S, b1[i].HP, b1[i].WP);
    }
    struct C2 { int K, S, H2, W2; size_t off; };
    const C2 b2[4] = {{5,3,169,20,WS_R2_4}, {3,2,126,14,WS_R2_8},
                      {1,1,126,14,WS_R2_16}, {1,1,62,6,WS_R2_32}};
    for (int i = 0; i < 4; ++i) {
        int total = Bsz * 3 * b2[i].H2 * b2[i].W2;
        conv2_bn_relu<<<(total + 255)/256, 256, 0, stream>>>(
            (const float*)(ws + b1[i].off), c2w[i], c2b[i],
            bn3g, bn3b, bn3m, bn3v, (float*)(ws + b2[i].off),
            b1[i].HP, b1[i].WP, b2[i].K, b2[i].S, b2[i].H2, b2[i].W2);
    }
    struct RZ { int Ho, Wo, seg; };
    const RZ rz[4] = {{90,9,0}, {67,6,2430}, {67,6,3636}, {33,2,4842}};
    float* FEATS = (float*)(ws + WS_FEATS);
    float* RFEAT = (float*)(ws + WS_RFEAT);
    for (int i = 0; i < 4; ++i) {
        int total = Bsz * 3 * rz[i].Ho * rz[i].Wo;
        resize_to_feats<<<(total + 255)/256, 256, 0, stream>>>(
            (const float*)(ws + b2[i].off), lengths, FEATS,
            b2[i].H2, b2[i].W2, rz[i].Ho, rz[i].Wo, rz[i].seg);
    }
    fc1_kernel<<<(Bsz*512 + 255)/256, 256, 0, stream>>>(FEATS, fc1w, fc1b, RFEAT);
    seq_build<<<(Bsz*Tsz*640 + 255)/256, 256, 0, stream>>>(batch, RFEAT, SEQ);

    // ---- 3) layer 0: input-projection GEMMs + bidirectional scan -----------
    const int Mrows = Bsz * Tsz;                     // 16384
    dim3 gg(Mrows / 64, GW / 64);                    // (256, 12)
    gemm_bf16<<<gg, 128, 0, stream>>>(SEQ, WIH0F, g0f_bih, XWF, Mrows, GW, 640);
    gemm_bf16<<<gg, 128, 0, stream>>>(SEQ, WIH0B, g0b_bih, XWB, Mrows, GW, 640);
    gru_scan<<<2, 384, 0, stream>>>(XWF, XWB, WHT0F, WHT0B, g0f_bhh, g0b_bhh, lengths, H0);

    // ---- 4) layer 1 ---------------------------------------------------------
    pack_bf16<<<(Mrows*512 + 255)/256, 256, 0, stream>>>(H0, H0BF, Mrows*512);
    gemm_bf16<<<gg, 128, 0, stream>>>(H0BF, WIH1F, g1f_bih, XWF, Mrows, GW, 512);
    gemm_bf16<<<gg, 128, 0, stream>>>(H0BF, WIH1B, g1b_bih, XWB, Mrows, GW, 512);
    gru_scan<<<2, 384, 0, stream>>>(XWF, XWB, WHT1F, WHT1B, g1f_bhh, g1b_bhh, lengths, H0);

    // ---- 5) final FC + length gather ---------------------------------------
    final_fc_gather<<<Bsz * Tsz, 64, 0, stream>>>(H0, fcw, fcb, lengths, (float*)d_out);
}
